// MaskedViT_70205535420656
// MI455X (gfx1250) — matrix operand order repaired and checked by reading, not compiled, not run on hardware
//
#include <hip/hip_runtime.h>
#include <hip/hip_bf16.h>
#include <math.h>

// ---------------------------------------------------------------------------
// MaskedViT forward for gfx1250 (MI455X). All big GEMMs use
// v_wmma_f32_16x16x32_f16 (f32 accumulate); LN/softmax/patch-embed are VALU.
// ---------------------------------------------------------------------------

#define DIM   768
#define LSEQ  197
#define NPAT  196
#define NHEAD 12
#define HDIM  64
#define DEPTH 12
#define BATCH 32
#define MROWS (BATCH * LSEQ)          // 6304

typedef __attribute__((ext_vector_type(16))) _Float16 v16h;
typedef __attribute__((ext_vector_type(8)))  float    v8f;

union FragU { v16h v; unsigned u[8]; };
union PackU { _Float16 h[4]; unsigned u[2]; };

// ---------------------------------------------------------------------------
// WMMA GEMM: Out[M,N] = epilogue(A[M,K] @ W[N,K]^T + bias[N])
//   op = 0: + bias                  op = 1: + bias + resid[M,N]
//   op = 2: gelu(x + bias)  (exact erf gelu)
// Block: 256 threads = 8 wave32, tile 64x64, K-step 64 (2 WMMA chunks).
// Pipelined: next K-tile's global_load_b128 are issued while the current
// tile's WMMAs consume LDS; cvt+ds_store happen one iteration later, so the
// load wait is overlapped with compute instead of serializing each load.
// ---------------------------------------------------------------------------
#define ASTR 72   // LDS row stride in f16 (multiple of 4 -> 8B-aligned quads)

__global__ __launch_bounds__(256) void gemm_wmma(
    const float* __restrict__ A, const float* __restrict__ W,
    const float* __restrict__ bias, const float* __restrict__ resid,
    float* __restrict__ Out, int M, int N, int K, int op)
{
    __shared__ _Float16 As[64 * ASTR];
    __shared__ _Float16 Ws[64 * ASTR];

    const int tid  = threadIdx.x;
    const int lane = tid & 31;
    const int wave = tid >> 5;
    const int wm   = wave >> 1;      // 0..3  (16-row strip)
    const int wn   = wave & 1;       // 0..1  (32-col strip)
    const int tileM = blockIdx.y * 64;
    const int tileN = blockIdx.x * 64;

    v8f c0 = {};
    v8f c1 = {};

    // staging geometry: 64 rows x 64 K as 1024 quads; 256 threads x 4 quads
    // quad idx -> row = idx>>4, kq = (idx&15)*4
    int ldsr[4], kq[4];
    float mskA[4];
    const float* aPtr[4];   // per-quad A source (clamped row, starts at k=kq)
    const float* wPtr[4];   // per-quad W source
    #pragma unroll
    for (int e = 0; e < 4; ++e) {
        int idx  = tid + e * 256;
        int r    = idx >> 4;
        ldsr[e]  = r;
        kq[e]    = (idx & 15) * 4;
        int grow = tileM + r;
        mskA[e]  = (grow < M) ? 1.0f : 0.0f;
        int rc   = (grow < M) ? grow : (M - 1);      // clamped, in-bounds
        aPtr[e]  = A + (size_t)rc * K + kq[e];
        wPtr[e]  = W + (size_t)(tileN + r) * K + kq[e];
    }

    // prologue: load first K-tile into registers
    float4 aReg[4], wReg[4];
    #pragma unroll
    for (int e = 0; e < 4; ++e) {
        aReg[e] = *(const float4*)(aPtr[e]);
        wReg[e] = *(const float4*)(wPtr[e]);
    }

    for (int k0 = 0; k0 < K; k0 += 64) {
        // convert + store the registers staged last iteration
        #pragma unroll
        for (int e = 0; e < 4; ++e) {
            PackU pa;
            pa.h[0] = (_Float16)(aReg[e].x * mskA[e]);
            pa.h[1] = (_Float16)(aReg[e].y * mskA[e]);
            pa.h[2] = (_Float16)(aReg[e].z * mskA[e]);
            pa.h[3] = (_Float16)(aReg[e].w * mskA[e]);
            *(unsigned*)&As[ldsr[e] * ASTR + kq[e]]     = pa.u[0];
            *(unsigned*)&As[ldsr[e] * ASTR + kq[e] + 2] = pa.u[1];
            PackU pw;
            pw.h[0] = (_Float16)wReg[e].x;
            pw.h[1] = (_Float16)wReg[e].y;
            pw.h[2] = (_Float16)wReg[e].z;
            pw.h[3] = (_Float16)wReg[e].w;
            *(unsigned*)&Ws[ldsr[e] * ASTR + kq[e]]     = pw.u[0];
            *(unsigned*)&Ws[ldsr[e] * ASTR + kq[e] + 2] = pw.u[1];
        }
        __syncthreads();

        // issue next tile's loads now; wait happens next iteration's cvt phase
        if (k0 + 64 < K) {
            #pragma unroll
            for (int e = 0; e < 4; ++e) {
                aReg[e] = *(const float4*)(aPtr[e] + k0 + 64);
                wReg[e] = *(const float4*)(wPtr[e] + k0 + 64);
            }
            if (k0 + 128 < K) {
                __builtin_prefetch(aPtr[0] + k0 + 128, 0, 1);
                __builtin_prefetch(wPtr[0] + k0 + 128, 0, 1);
            }
        }

        const int hi = lane >> 4;
        const int rowoff = (wm * 16 + (lane & 15)) * ASTR;
        const int n0off  = (wn * 32 + (lane & 15)) * ASTR;
        const int n1off  = n0off + 16 * ASTR;
        #pragma unroll
        for (int kc = 0; kc < 2; ++kc) {
            const int kcb = kc * 32;
            // A fragment: ISA 16-bit A layout (lane = M row)
            FragU a;
            #pragma unroll
            for (int v = 0; v < 8; ++v) {
                int kb = ((v < 4) ? 0 : 16) + hi * 8 + (v & 3) * 2;
                a.u[v] = *(const unsigned*)&As[rowoff + kcb + kb];
            }
            // B fragments: same K order as A (lanes 0-15 K=0..7,16..23; lanes 16-31 K=8..15,24..31)
            FragU b0, b1;
            #pragma unroll
            for (int v = 0; v < 8; ++v) {
                int kb = ((v < 4) ? 0 : 16) + hi * 8 + (v & 3) * 2;
                b0.u[v] = *(const unsigned*)&Ws[n0off + kcb + kb];
                b1.u[v] = *(const unsigned*)&Ws[n1off + kcb + kb];
            }
            c0 = __builtin_amdgcn_wmma_f32_16x16x32_f16(false, a.v, false, b0.v,
                                                        (short)0, c0, false, false);
            c1 = __builtin_amdgcn_wmma_f32_16x16x32_f16(false, a.v, false, b1.v,
                                                        (short)0, c1, false, false);
        }
        __syncthreads();
    }

    // epilogue: C/D layout — VGPR r holds M = r + 8*(lane>>4), N = lane&15
    const int hi = lane >> 4;
    const int ncol0 = tileN + wn * 32 + (lane & 15);
    const int ncol1 = ncol0 + 16;
    #pragma unroll
    for (int r = 0; r < 8; ++r) {
        int grow = tileM + wm * 16 + 8 * hi + r;
        if (grow >= M) continue;
        float v0 = c0[r] + bias[ncol0];
        float v1 = c1[r] + bias[ncol1];
        if (op == 1) {
            v0 += resid[(size_t)grow * N + ncol0];
            v1 += resid[(size_t)grow * N + ncol1];
        } else if (op == 2) {
            v0 = 0.5f * v0 * (1.0f + erff(v0 * 0.70710678118f));
            v1 = 0.5f * v1 * (1.0f + erff(v1 * 0.70710678118f));
        }
        Out[(size_t)grow * N + ncol0] = v0;
        Out[(size_t)grow * N + ncol1] = v1;
    }
}

// ---------------------------------------------------------------------------
// Row layernorm over 768 features: out = g*(x-m)*rsqrt(v+eps)+b
// ---------------------------------------------------------------------------
__global__ __launch_bounds__(256) void ln_rows(
    const float* __restrict__ in, const float* __restrict__ g,
    const float* __restrict__ bta, float* __restrict__ out, int Mrows)
{
    int row = blockIdx.x;
    if (row >= Mrows) return;
    const float* x = in + (size_t)row * DIM;
    __shared__ float red[256];
    int tid = threadIdx.x;

    float s = 0.0f;
    for (int d = tid; d < DIM; d += 256) s += x[d];
    red[tid] = s; __syncthreads();
    for (int o = 128; o > 0; o >>= 1) {
        if (tid < o) red[tid] += red[tid + o];
        __syncthreads();
    }
    float mean = red[0] * (1.0f / DIM);
    __syncthreads();

    float v = 0.0f;
    for (int d = tid; d < DIM; d += 256) { float t = x[d] - mean; v += t * t; }
    red[tid] = v; __syncthreads();
    for (int o = 128; o > 0; o >>= 1) {
        if (tid < o) red[tid] += red[tid + o];
        __syncthreads();
    }
    float rstd = rsqrtf(red[0] * (1.0f / DIM) + 1e-5f);

    for (int d = tid; d < DIM; d += 256)
        out[(size_t)row * DIM + d] = g[d] * (x[d] - mean) * rstd + bta[d];
}

// ---------------------------------------------------------------------------
// Patch embed + LN: one block per (b, patch); C = 3 (img) or 1 (mask)
// ---------------------------------------------------------------------------
__global__ __launch_bounds__(256) void patch_embed(
    const float* __restrict__ in, const float* __restrict__ w,
    const float* __restrict__ bvec, const float* __restrict__ g,
    const float* __restrict__ beta, float* __restrict__ tok, int C)
{
    int bp = blockIdx.x;
    int b = bp / NPAT, p = bp % NPAT;
    int py = p / 14, px = p % 14;
    int KK = C * 256;

    __shared__ float patch[768];
    __shared__ float red[256];
    int tid = threadIdx.x;

    for (int kk = tid; kk < KK; kk += 256) {
        int c = kk >> 8, rem = kk & 255, iy = rem >> 4, ix = rem & 15;
        patch[kk] = in[(((size_t)b * C + c) * 224 + py * 16 + iy) * 224 + px * 16 + ix];
    }
    __syncthreads();

    float t[3];
    #pragma unroll
    for (int e = 0; e < 3; ++e) {
        int n = tid + e * 256;
        float acc = bvec[n];
        const float* wr = w + (size_t)n * KK;
        for (int kk = 0; kk < KK; ++kk) acc += patch[kk] * wr[kk];
        t[e] = acc;
    }
    __syncthreads();

    red[tid] = t[0] + t[1] + t[2];
    __syncthreads();
    for (int o = 128; o > 0; o >>= 1) {
        if (tid < o) red[tid] += red[tid + o];
        __syncthreads();
    }
    float mean = red[0] * (1.0f / DIM);
    __syncthreads();
    float vv = 0.0f;
    #pragma unroll
    for (int e = 0; e < 3; ++e) { float d = t[e] - mean; vv += d * d; }
    red[tid] = vv; __syncthreads();
    for (int o = 128; o > 0; o >>= 1) {
        if (tid < o) red[tid] += red[tid + o];
        __syncthreads();
    }
    float rstd = rsqrtf(red[0] * (1.0f / DIM) + 1e-5f);

    #pragma unroll
    for (int e = 0; e < 3; ++e) {
        int n = tid + e * 256;
        tok[(size_t)bp * DIM + n] = g[n] * (t[e] - mean) * rstd + beta[n];
    }
}

// zero-fill (for bias padding row/col)
__global__ __launch_bounds__(256) void fill_zero(float* __restrict__ p, int n)
{
    int i = blockIdx.x * 256 + threadIdx.x;
    if (i < n) p[i] = 0.0f;
}

// bias[b, i+1, j+1] = <tok_m[b,i], tok_m[b,j]> / sqrt(D); one block per (b,i)
__global__ __launch_bounds__(256) void bias_outer(
    const float* __restrict__ tokm, float* __restrict__ bias)
{
    int bp = blockIdx.x;
    int b = bp / NPAT, i = bp % NPAT;
    __shared__ float rowi[DIM];
    int tid = threadIdx.x;
    for (int d = tid; d < DIM; d += 256)
        rowi[d] = tokm[((size_t)b * NPAT + i) * DIM + d];
    __syncthreads();
    const float scale = 0.036084392f; // 1/sqrt(768)
    for (int j = tid; j < NPAT; j += 256) {
        const float* rj = tokm + ((size_t)b * NPAT + j) * DIM;
        float acc = 0.0f;
        for (int d = 0; d < DIM; ++d) acc += rowi[d] * rj[d];
        bias[((size_t)b * LSEQ + (i + 1)) * LSEQ + (j + 1)] = acc * scale;
    }
}

// x[b,0,:] = cls + pos[0]; x[b,1+l,:] = tok_i + 0.5*tok_m + pos[1+l]
__global__ __launch_bounds__(256) void assemble_x(
    const float* __restrict__ toki, const float* __restrict__ tokm,
    const float* __restrict__ cls, const float* __restrict__ pos,
    float* __restrict__ x)
{
    int idx = blockIdx.x * 256 + threadIdx.x;
    const int total = MROWS * DIM;
    if (idx >= total) return;
    int d = idx % DIM;
    int bl = idx / DIM;
    int l = bl % LSEQ;
    int b = bl / LSEQ;
    float v;
    if (l == 0) {
        v = cls[d] + pos[d];
    } else {
        size_t t = ((size_t)b * NPAT + (l - 1)) * DIM + d;
        v = toki[t] + 0.5f * tokm[t] + pos[(size_t)l * DIM + d];
    }
    x[idx] = v;
}

// ---------------------------------------------------------------------------
// Attention: one block per (head, batch). K/V staged in LDS as f16.
// qkv layout: [B*L, 3*D]; q at +0, k at +768, v at +1536 (head h -> h*64).
// ---------------------------------------------------------------------------
__global__ __launch_bounds__(256) void attn_kernel(
    const float* __restrict__ qkv, const float* __restrict__ bias,
    float* __restrict__ o)
{
    int head = blockIdx.x, b = blockIdx.y;
    __shared__ _Float16 Ks[LSEQ * HDIM];
    __shared__ _Float16 Vs[LSEQ * HDIM];
    __shared__ float prob[8][204];

    int tid = threadIdx.x, lane = tid & 31, wave = tid >> 5;

    for (int idx = tid; idx < LSEQ * HDIM; idx += 256) {
        int j = idx >> 6, d = idx & 63;
        size_t base = ((size_t)b * LSEQ + j) * (3 * DIM) + head * HDIM + d;
        Ks[idx] = (_Float16)qkv[base + DIM];
        Vs[idx] = (_Float16)qkv[base + 2 * DIM];
    }
    __syncthreads();

    for (int i = wave; i < LSEQ; i += 8) {
        size_t qbase = ((size_t)b * LSEQ + i) * (3 * DIM) + head * HDIM;
        float qv[HDIM];
        #pragma unroll
        for (int d = 0; d < HDIM; ++d) qv[d] = qkv[qbase + d];

        float s[7];
        float mx = -1e30f;
        #pragma unroll
        for (int t = 0; t < 7; ++t) {
            int j = t * 32 + lane;
            float acc = -1e30f;
            if (j < LSEQ) {
                acc = 0.0f;
                for (int d = 0; d < HDIM; ++d)
                    acc += qv[d] * (float)Ks[j * HDIM + d];
                acc = acc * 0.125f +
                      bias[((size_t)b * LSEQ + i) * LSEQ + j];
            }
            s[t] = acc;
            mx = fmaxf(mx, acc);
        }
        for (int off = 16; off > 0; off >>= 1)
            mx = fmaxf(mx, __shfl_xor(mx, off, 32));

        float sum = 0.0f;
        #pragma unroll
        for (int t = 0; t < 7; ++t) {
            int j = t * 32 + lane;
            float p = (j < LSEQ) ? expf(s[t] - mx) : 0.0f;
            s[t] = p;
            sum += p;
        }
        for (int off = 16; off > 0; off >>= 1)
            sum += __shfl_xor(sum, off, 32);
        float inv = 1.0f / sum;

        #pragma unroll
        for (int t = 0; t < 7; ++t) {
            int j = t * 32 + lane;
            if (j < LSEQ) prob[wave][j] = s[t] * inv;
        }
        __builtin_amdgcn_wave_barrier();   // per-wave LDS exchange; no WG barrier

        float a0 = 0.0f, a1 = 0.0f;
        for (int j = 0; j < LSEQ; ++j) {
            float p = prob[wave][j];
            a0 += p * (float)Vs[j * HDIM + lane];
            a1 += p * (float)Vs[j * HDIM + lane + 32];
        }
        size_t obase = ((size_t)b * LSEQ + i) * DIM + head * HDIM;
        o[obase + lane]      = a0;
        o[obase + lane + 32] = a1;
        __builtin_amdgcn_wave_barrier();
    }
}

// head: out[b, c] = <x[b,0,:], head_w[c]> + head_b[c]
__global__ __launch_bounds__(256) void head_kernel(
    const float* __restrict__ x, const float* __restrict__ hw,
    const float* __restrict__ hb, float* __restrict__ out)
{
    int b = blockIdx.x;
    const float* xr = x + (size_t)b * LSEQ * DIM;   // row 0 of batch b
    __shared__ float red[4][256];
    int tid = threadIdx.x;
    float a[4] = {0, 0, 0, 0};
    for (int d = tid; d < DIM; d += 256) {
        float xv = xr[d];
        #pragma unroll
        for (int c = 0; c < 4; ++c) a[c] += xv * hw[c * DIM + d];
    }
    #pragma unroll
    for (int c = 0; c < 4; ++c) red[c][tid] = a[c];
    __syncthreads();
    for (int o = 128; o > 0; o >>= 1) {
        if (tid < o)
            #pragma unroll
            for (int c = 0; c < 4; ++c) red[c][tid] += red[c][tid + o];
        __syncthreads();
    }
    if (tid < 4) out[b * 4 + tid] = red[tid][0] + hb[tid];
}

// ---------------------------------------------------------------------------
// Host-side orchestration
// ---------------------------------------------------------------------------
extern "C" void kernel_launch(void* const* d_in, const int* in_sizes, int n_in,
                              void* d_out, int out_size, void* d_ws, size_t ws_size,
                              hipStream_t stream)
{
    (void)in_sizes; (void)n_in; (void)out_size; (void)ws_size;

    const float* img      = (const float*)d_in[0];
    const float* mask     = (const float*)d_in[1];
    const float* pe_img_w = (const float*)d_in[2];
    const float* pe_img_b = (const float*)d_in[3];
    const float* pe_img_g = (const float*)d_in[4];
    const float* pe_img_bb= (const float*)d_in[5];
    const float* pe_msk_w = (const float*)d_in[6];
    const float* pe_msk_b = (const float*)d_in[7];
    const float* pe_msk_g = (const float*)d_in[8];
    const float* pe_msk_bb= (const float*)d_in[9];
    const float* cls      = (const float*)d_in[10];
    const float* pos      = (const float*)d_in[11];
    const float* ln1_g    = (const float*)d_in[12];
    const float* ln1_b    = (const float*)d_in[13];
    const float* w_qkv    = (const float*)d_in[14];
    const float* b_qkv    = (const float*)d_in[15];
    const float* w_out    = (const float*)d_in[16];
    const float* b_out    = (const float*)d_in[17];
    const float* ln2_g    = (const float*)d_in[18];
    const float* ln2_b    = (const float*)d_in[19];
    const float* w_fc1    = (const float*)d_in[20];
    const float* b_fc1    = (const float*)d_in[21];
    const float* w_fc2    = (const float*)d_in[22];
    const float* b_fc2    = (const float*)d_in[23];
    const float* enc_g    = (const float*)d_in[24];
    const float* enc_b    = (const float*)d_in[25];
    const float* fin_g    = (const float*)d_in[26];
    const float* fin_b    = (const float*)d_in[27];
    const float* head_w   = (const float*)d_in[28];
    const float* head_b   = (const float*)d_in[29];
    float* out = (float*)d_out;

    // workspace carve (floats)
    const size_t XE  = (size_t)MROWS * DIM;             // 4,841,472
    const size_t GE  = (size_t)MROWS * 4 * DIM;         // 19,365,888 (>= qkv 3D)
    const size_t TME = (size_t)BATCH * NPAT * DIM;      // 4,816,896
    const size_t BE  = (size_t)BATCH * LSEQ * LSEQ;     // 1,241,888
    float* ws  = (float*)d_ws;
    float* X    = ws;
    float* Hb   = X + XE;        // ln output / attn output / tok_i
    float* G    = Hb + XE;       // qkv & MLP hidden & final-LN temp
    float* TOKM = G + GE;
    float* BIAS = TOKM + TME;
    float* TOKI = Hb;            // alias: tok_i lives in Hb until assemble

    // 1) patch embeds (+LN)
    patch_embed<<<BATCH * NPAT, 256, 0, stream>>>(
        img,  pe_img_w, pe_img_b, pe_img_g, pe_img_bb, TOKI, 3);
    patch_embed<<<BATCH * NPAT, 256, 0, stream>>>(
        mask, pe_msk_w, pe_msk_b, pe_msk_g, pe_msk_bb, TOKM, 1);

    // 2) attention bias (zero pad row/col, then interior)
    fill_zero<<<(int)((BE + 255) / 256), 256, 0, stream>>>(BIAS, (int)BE);
    bias_outer<<<BATCH * NPAT, 256, 0, stream>>>(TOKM, BIAS);

    // 3) assemble token sequence
    assemble_x<<<(int)((XE + 255) / 256), 256, 0, stream>>>(TOKI, TOKM, cls, pos, X);

    // 4) transformer blocks
    const int MT = (MROWS + 63) / 64;   // 99 M-tiles
    for (int l = 0; l < DEPTH; ++l) {
        const float* l1g = ln1_g + (size_t)l * DIM;
        const float* l1b = ln1_b + (size_t)l * DIM;
        const float* wq  = w_qkv + (size_t)l * 3 * DIM * DIM;
        const float* bq  = b_qkv + (size_t)l * 3 * DIM;
        const float* wo  = w_out + (size_t)l * DIM * DIM;
        const float* bo  = b_out + (size_t)l * DIM;
        const float* l2g = ln2_g + (size_t)l * DIM;
        const float* l2b = ln2_b + (size_t)l * DIM;
        const float* w1  = w_fc1 + (size_t)l * 4 * DIM * DIM;
        const float* b1  = b_fc1 + (size_t)l * 4 * DIM;
        const float* w2  = w_fc2 + (size_t)l * DIM * 4 * DIM;
        const float* b2  = b_fc2 + (size_t)l * DIM;

        ln_rows<<<MROWS, 256, 0, stream>>>(X, l1g, l1b, Hb, MROWS);
        gemm_wmma<<<dim3(3 * DIM / 64, MT), 256, 0, stream>>>(
            Hb, wq, bq, nullptr, G, MROWS, 3 * DIM, DIM, 0);          // qkv
        attn_kernel<<<dim3(NHEAD, BATCH), 256, 0, stream>>>(G, BIAS, Hb);
        gemm_wmma<<<dim3(DIM / 64, MT), 256, 0, stream>>>(
            Hb, wo, bo, X, X, MROWS, DIM, DIM, 1);                    // out proj + res
        ln_rows<<<MROWS, 256, 0, stream>>>(X, l2g, l2b, Hb, MROWS);
        gemm_wmma<<<dim3(4 * DIM / 64, MT), 256, 0, stream>>>(
            Hb, w1, b1, nullptr, G, MROWS, 4 * DIM, DIM, 2);          // fc1 + gelu
        gemm_wmma<<<dim3(DIM / 64, MT), 256, 0, stream>>>(
            G, w2, b2, X, X, MROWS, DIM, 4 * DIM, 1);                 // fc2 + res
    }

    // 5) final norms + head (LN is per-row; head reads only CLS rows)
    ln_rows<<<MROWS, 256, 0, stream>>>(X,  enc_g, enc_b, Hb, MROWS);
    ln_rows<<<MROWS, 256, 0, stream>>>(Hb, fin_g, fin_b, G,  MROWS);
    head_kernel<<<BATCH, 256, 0, stream>>>(G, head_w, head_b, out);
}
